// GAT_10213432229980
// MI455X (gfx1250) — matrix operand order repaired and checked by reading, not compiled, not run on hardware
//
#include <hip/hip_runtime.h>
#include <hip/hip_bf16.h>
#include <math.h>

typedef _Float16 v16h __attribute__((ext_vector_type(16)));
typedef _Float16 v8h  __attribute__((ext_vector_type(8)));
typedef float    v8f  __attribute__((ext_vector_type(8)));

#define GN   50000
#define GE   800000
#define GET  850000      // E + N self loops
#define GF   128
#define NHD  8
#define CH1  16
#define NC   40
#define NCP  48          // 40 padded to 3 WMMA n-tiles
#define MT   3125        // 50000 / 16 m-tiles
#define GWAVES 4         // waves per GEMM block
#define NEG_SLOPE 0.2f

#define CDIV(a,b) (((a)+(b)-1)/(b))

// ---------------- scratch layout (bytes, all 256-aligned) ----------------
static const size_t o_xh  = 0;          // x f16 [N,128]; later H1e f16 [N,128]
static const size_t o_w1h = 12800000;   // W1^T f16 [128 cols][128 k]
static const size_t o_w2h = 12832768;   // W2^T f16 [48 cols][128 k], cols>=40 zero
static const size_t o_H1  = 12845056;   // H1 f32 [N,128]; later H2 f32 [N,40]
static const size_t o_as1 = 38445056;   // [N,8]; later [N]
static const size_t o_ad1 = 40045056;   // [N,8]; later [N]
static const size_t o_dn1 = 41645056;   // [N,8]; later [N]
static const size_t o_ex1 = 43245056;   // [ET,8]; later [ET]
static const size_t o_ac1 = 70445056;   // [N,128]; later [N,40]
// total required: 96,045,056 bytes (~92 MiB)

// ---------------- conversion kernels ----------------
__global__ void cvt_x_f16(const float* __restrict__ x, _Float16* __restrict__ xh) {
    int i = blockIdx.x * blockDim.x + threadIdx.x;
    if (i < GN * GF) xh[i] = (_Float16)x[i];
}

__global__ void cvt_w_f16(const float* __restrict__ W1, const float* __restrict__ W2,
                          _Float16* __restrict__ w1t, _Float16* __restrict__ w2t) {
    int i = blockIdx.x * blockDim.x + threadIdx.x;
    if (i < GF * GF) {                       // W1^T: [col][k]
        int k = i >> 7, c = i & 127;
        w1t[(size_t)c * GF + k] = (_Float16)W1[(size_t)k * GF + c];
    } else if (i < GF * GF + NCP * GF) {     // W2^T padded: [col(48)][k]
        int t = i - GF * GF;
        int k = t & 127, c = t >> 7;         // c in 0..47
        float v = (c < NC) ? W2[(size_t)k * NC + c] : 0.f;
        w2t[(size_t)c * GF + k] = (_Float16)v;
    }
}

// ---------------- WMMA GEMM, K=128 fixed ----------------
// A: [Mtiles*16, 128] f16 row-major.  BT: [Ntiles*16 cols][128 k] f16 (LDS-staged).
// D: [Mtiles*16, ldd] f32; store only cols < storeCols (storeCols % 8 == 0).
__global__ __launch_bounds__(GWAVES * 32) void gemm_wmma_k128(
    const _Float16* __restrict__ A, const _Float16* __restrict__ BT,
    float* __restrict__ D, int Mtiles, int Ntiles, int ldd, int storeCols)
{
    __shared__ __align__(16) _Float16 Bs[8 * 16 * GF];   // up to 8 n-tiles: 32 KB
    __shared__ __align__(16) float    Cs[GWAVES * 256];  // per-wave transpose tiles

    int tid  = threadIdx.x;
    int wave = tid >> 5, lane = tid & 31;
    int m = lane & 15, g = lane >> 4;

    // cooperative stage of BT into LDS (16-byte chunks)
    int nchunks = (Ntiles * 16 * GF) >> 3;               // halves / 8
    const uint4* bsrc = (const uint4*)BT;
    uint4*       bdst = (uint4*)Bs;
    for (int i = tid; i < nchunks; i += GWAVES * 32) bdst[i] = bsrc[i];
    __syncthreads();

    int mt = blockIdx.x * GWAVES + wave;
    if (mt < Mtiles) {                                   // wave-uniform guard
        // Load the 4 A fragments for this 16-row strip (kept across n-tiles).
        v16h afrag[4];
        const _Float16* abase = A + ((size_t)(mt * 16 + m)) * GF;
#pragma unroll
        for (int kt = 0; kt < 4; ++kt) {
            const _Float16* ap = abase + kt * 32;
            v8h lo = *(const v8h*)(ap + g * 8);          // K = g*8 .. g*8+7
            v8h hi = *(const v8h*)(ap + 16 + g * 8);     // K = 16+g*8 ..
            v16h af;
#pragma unroll
            for (int i = 0; i < 8; ++i) { af[i] = lo[i]; af[8 + i] = hi[i]; }
            afrag[kt] = af;
        }

        float* tile = &Cs[wave * 256];
        for (int nt = 0; nt < Ntiles; ++nt) {
            v8f acc = {};
            const _Float16* bbase = &Bs[(nt * 16 + m) * GF];
#pragma unroll
            for (int kt = 0; kt < 4; ++kt) {
                const _Float16* bp = bbase + kt * 32 + g * 8; // K = g*8.. and 16+g*8..
                v8h blo = *(const v8h*)(bp);
                v8h bhi = *(const v8h*)(bp + 16);
                v16h bf;
#pragma unroll
                for (int i = 0; i < 8; ++i) { bf[i] = blo[i]; bf[8 + i] = bhi[i]; }
                acc = __builtin_amdgcn_wmma_f32_16x16x32_f16(
                    false, afrag[kt], false, bf, (short)0, acc, false, false);
            }
            // transpose tile through LDS, then coalesced b128 stores
#pragma unroll
            for (int r = 0; r < 8; ++r) tile[(g * 8 + r) * 16 + m] = acc[r];
            asm volatile("s_wait_dscnt 0x0" ::: "memory");
            int row     = mt * 16 + (lane >> 1);
            int colbase = nt * 16 + (lane & 1) * 8;
            if (colbase < storeCols) {
                const float4* t4 = (const float4*)&tile[(lane >> 1) * 16 + (lane & 1) * 8];
                float4* d4 = (float4*)&D[(size_t)row * ldd + colbase];
                d4[0] = t4[0];
                d4[1] = t4[1];
            }
        }
    }
}

// ---------------- layer-1 attention logits ----------------
__global__ void att_logits1(const float* __restrict__ H1,
                            const float* __restrict__ att_src,
                            const float* __restrict__ att_dst,
                            float* __restrict__ as, float* __restrict__ ad) {
    int i = blockIdx.x * blockDim.x + threadIdx.x;
    if (i >= GN * NHD) return;
    int n = i >> 3, h = i & 7;
    const float* hp = H1 + (size_t)n * GF + h * CH1;
    float s = 0.f, d = 0.f;
#pragma unroll
    for (int c = 0; c < CH1; ++c) {
        float v = hp[c];
        s += v * att_src[h * CH1 + c];
        d += v * att_dst[h * CH1 + c];
    }
    as[i] = s; ad[i] = d;
}

__device__ __forceinline__ void edge_sd(const long long* __restrict__ ei, int e,
                                        int& s, int& d) {
    if (e < GE) { s = (int)ei[e]; d = (int)ei[GE + e]; }
    else        { s = e - GE; d = e - GE; }   // appended self loop
}

// pass A: ex = exp(leakyrelu(a_src[s]+a_dst[d])); denom[d] += ex
__global__ void edge1_a(const long long* __restrict__ ei,
                        const float* __restrict__ as, const float* __restrict__ ad,
                        float* __restrict__ ex, float* __restrict__ denom) {
    int e = blockIdx.x * blockDim.x + threadIdx.x;
    if (e >= GET) return;
    int s, d; edge_sd(ei, e, s, d);
#pragma unroll
    for (int h = 0; h < NHD; ++h) {
        float al = as[s * NHD + h] + ad[d * NHD + h];
        al = (al > 0.f) ? al : NEG_SLOPE * al;
        float v = expf(al);
        ex[(size_t)e * NHD + h] = v;
        atomicAdd(&denom[d * NHD + h], v);
    }
}

// pass B: accum[d] += h[s] * ex/denom[d]
__global__ void edge1_b(const long long* __restrict__ ei,
                        const float* __restrict__ ex, const float* __restrict__ denom,
                        const float* __restrict__ H1, float* __restrict__ accum) {
    int i = blockIdx.x * blockDim.x + threadIdx.x;
    if (i >= GET * NHD) return;
    int e = i >> 3, h = i & 7;
    int s, d; edge_sd(ei, e, s, d);
    float coef = ex[(size_t)e * NHD + h] / denom[d * NHD + h];
    const float* hr = H1 + (size_t)s * GF + h * CH1;
    float* orow = accum + (size_t)d * GF + h * CH1;
#pragma unroll
    for (int c = 0; c < CH1; ++c) atomicAdd(&orow[c], hr[c] * coef);
}

// +b1, ELU, convert to f16 for GEMM2
__global__ void finalize1(const float* __restrict__ accum, const float* __restrict__ b1,
                          _Float16* __restrict__ h1e) {
    int i = blockIdx.x * blockDim.x + threadIdx.x;
    if (i >= GN * GF) return;
    float v = accum[i] + b1[i & 127];
    v = (v > 0.f) ? v : (expf(v) - 1.f);      // ELU, alpha=1
    h1e[i] = (_Float16)v;
}

// ---------------- layer-2 attention ----------------
__global__ void att_logits2(const float* __restrict__ H2,
                            const float* __restrict__ att_src,
                            const float* __restrict__ att_dst,
                            float* __restrict__ as, float* __restrict__ ad) {
    int n = blockIdx.x * blockDim.x + threadIdx.x;
    if (n >= GN) return;
    const float* hp = H2 + (size_t)n * NC;
    float s = 0.f, d = 0.f;
#pragma unroll
    for (int c = 0; c < NC; ++c) {
        float v = hp[c];
        s += v * att_src[c];
        d += v * att_dst[c];
    }
    as[n] = s; ad[n] = d;
}

__global__ void edge2_a(const long long* __restrict__ ei,
                        const float* __restrict__ as, const float* __restrict__ ad,
                        float* __restrict__ ex, float* __restrict__ denom) {
    int e = blockIdx.x * blockDim.x + threadIdx.x;
    if (e >= GET) return;
    int s, d; edge_sd(ei, e, s, d);
    float al = as[s] + ad[d];
    al = (al > 0.f) ? al : NEG_SLOPE * al;
    float v = expf(al);
    ex[e] = v;
    atomicAdd(&denom[d], v);
}

__global__ void edge2_b(const long long* __restrict__ ei,
                        const float* __restrict__ ex, const float* __restrict__ denom,
                        const float* __restrict__ H2, float* __restrict__ accum) {
    int e = blockIdx.x * blockDim.x + threadIdx.x;
    if (e >= GET) return;
    int s, d; edge_sd(ei, e, s, d);
    float coef = ex[e] / denom[d];
    const float* hr = H2 + (size_t)s * NC;
    float* orow = accum + (size_t)d * NC;
#pragma unroll
    for (int c = 0; c < NC; ++c) atomicAdd(&orow[c], hr[c] * coef);
}

__global__ void finalize2(const float* __restrict__ accum, const float* __restrict__ b2,
                          float* __restrict__ out) {
    int i = blockIdx.x * blockDim.x + threadIdx.x;
    if (i >= GN * NC) return;
    out[i] = accum[i] + b2[i % NC];
}

// ---------------- host launch ----------------
extern "C" void kernel_launch(void* const* d_in, const int* in_sizes, int n_in,
                              void* d_out, int out_size, void* d_ws, size_t ws_size,
                              hipStream_t stream) {
    const float*     x        = (const float*)d_in[0];
    const long long* ei       = (const long long*)d_in[1];   // jnp.int64 [2,E]
    const float*     W1       = (const float*)d_in[2];
    const float*     att_src1 = (const float*)d_in[3];
    const float*     att_dst1 = (const float*)d_in[4];
    const float*     b1       = (const float*)d_in[5];
    const float*     W2       = (const float*)d_in[6];
    const float*     att_src2 = (const float*)d_in[7];
    const float*     att_dst2 = (const float*)d_in[8];
    const float*     b2       = (const float*)d_in[9];
    float* out = (float*)d_out;

    char* ws = (char*)d_ws;
    _Float16* xh   = (_Float16*)(ws + o_xh);
    _Float16* w1t  = (_Float16*)(ws + o_w1h);
    _Float16* w2t  = (_Float16*)(ws + o_w2h);
    float*    H1   = (float*)(ws + o_H1);
    float*    as1  = (float*)(ws + o_as1);
    float*    ad1  = (float*)(ws + o_ad1);
    float*    dn1  = (float*)(ws + o_dn1);
    float*    ex1  = (float*)(ws + o_ex1);
    float*    ac1  = (float*)(ws + o_ac1);
    // layer-2 reuse
    _Float16* h1e  = (_Float16*)(ws + o_xh);    // [N,128] f16 (xh dead)
    float*    H2   = (float*)(ws + o_H1);       // [N,40]  (H1 dead after edge1_b)
    float*    as2  = (float*)(ws + o_as1);
    float*    ad2  = (float*)(ws + o_ad1);
    float*    dn2  = (float*)(ws + o_dn1);
    float*    ex2  = (float*)(ws + o_ex1);
    float*    ac2  = (float*)(ws + o_ac1);

    const int B = 256;
    const int gemmBlocks = CDIV(MT, GWAVES);

    // ---- layer 1 ----
    cvt_x_f16<<<CDIV(GN * GF, B), B, 0, stream>>>(x, xh);
    cvt_w_f16<<<CDIV(GF * GF + NCP * GF, B), B, 0, stream>>>(W1, W2, w1t, w2t);

    gemm_wmma_k128<<<gemmBlocks, GWAVES * 32, 0, stream>>>(xh, w1t, H1, MT, GF / 16, GF, GF);

    att_logits1<<<CDIV(GN * NHD, B), B, 0, stream>>>(H1, att_src1, att_dst1, as1, ad1);

    hipMemsetAsync(dn1, 0, (size_t)GN * NHD * sizeof(float), stream);
    edge1_a<<<CDIV(GET, B), B, 0, stream>>>(ei, as1, ad1, ex1, dn1);

    hipMemsetAsync(ac1, 0, (size_t)GN * GF * sizeof(float), stream);
    edge1_b<<<CDIV(GET * NHD, B), B, 0, stream>>>(ei, ex1, dn1, H1, ac1);

    finalize1<<<CDIV(GN * GF, B), B, 0, stream>>>(ac1, b1, h1e);

    // ---- layer 2 ----
    gemm_wmma_k128<<<gemmBlocks, GWAVES * 32, 0, stream>>>(h1e, w2t, H2, MT, NCP / 16, NC, NC);

    att_logits2<<<CDIV(GN, B), B, 0, stream>>>(H2, att_src2, att_dst2, as2, ad2);

    hipMemsetAsync(dn2, 0, (size_t)GN * sizeof(float), stream);
    edge2_a<<<CDIV(GET, B), B, 0, stream>>>(ei, as2, ad2, ex2, dn2);

    hipMemsetAsync(ac2, 0, (size_t)GN * NC * sizeof(float), stream);
    edge2_b<<<CDIV(GET, B), B, 0, stream>>>(ei, ex2, dn2, H2, ac2);

    finalize2<<<CDIV(GN * NC, B), B, 0, stream>>>(ac2, b2, out);
}
